// LHKGNN_12506944766710
// MI455X (gfx1250) — matrix operand order repaired and checked by reading, not compiled, not run on hardware
//
#include <hip/hip_runtime.h>
#include <hip/hip_bf16.h>

// ---------------------------------------------------------------------------
// CDNA5 (gfx1250) implementation of the heat-kernel GNN reference.
//  - MLP GEMMs via v_wmma_f32_16x16x32_bf16 (bf16 in, f32 accum),
//    LDS-staged weight stripes shared by all 8 waves of a workgroup
//  - Dense L_tilde replaced by mathematically-identical sparse scatter SpMM
//  - Bessel/Chebyshev coefficients via stable term recurrence
// ---------------------------------------------------------------------------

typedef __attribute__((ext_vector_type(16))) __bf16   v16bf;
typedef __attribute__((ext_vector_type(8)))  float    v8f;
typedef __attribute__((ext_vector_type(4)))  unsigned v4u;   // trivial 128-bit load type

static constexpr int IN_DIM  = 512;
static constexpr int HIDDEN  = 256;
static constexpr int OUT_DIM = 64;
static constexpr int KCHEB   = 10;

static constexpr int KCHUNK  = 128;            // K halves staged in LDS per step
static constexpr int RPAD    = 8;              // row pad (halves) -> 272B stride, bank-conflict free
static constexpr int RSTRIDE = KCHUNK + RPAD;  // LDS row stride in halves

__device__ __forceinline__ unsigned short f2bf(float f) {
    unsigned u = __float_as_uint(f);
    u += 0x7FFFu + ((u >> 16) & 1u);        // round-to-nearest-even
    return (unsigned short)(u >> 16);
}

union FragBF {                               // both members trivial -> union OK
    v16bf bf;
    v4u   q[2];
};

// ---------------------------- utility kernels ------------------------------

__global__ void zero_f32_kernel(float* p, long n) {
    long i = (long)blockIdx.x * blockDim.x + threadIdx.x;
    if (i < n) p[i] = 0.0f;
}

__global__ void cvt_bf16_kernel(const float* __restrict__ src,
                                unsigned short* __restrict__ dst, long n) {
    long i = (long)blockIdx.x * blockDim.x + threadIdx.x;
    if (i < n) dst[i] = f2bf(src[i]);
}

// w [rowsK, cols] (row-major) -> wt [cols, rowsK] bf16 (so B-fragment K is contiguous)
__global__ void transpose_cvt_kernel(const float* __restrict__ w,
                                     unsigned short* __restrict__ wt,
                                     int rowsK, int cols) {
    long i = (long)blockIdx.x * blockDim.x + threadIdx.x;
    long total = (long)rowsK * cols;
    if (i >= total) return;
    int r = (int)(i / cols);
    int c = (int)(i % cols);
    wt[(long)c * rowsK + r] = f2bf(w[i]);
}

// ----------------------------- WMMA GEMM -----------------------------------
// Block = 256 threads = 8 waves. Each wave owns a 16-row tile; the block
// shares one 64-column stripe of B staged through LDS (4 accumulators/wave).
// A: [rowsA x Kdim] bf16 row-major. Bt: [cols x Kdim] bf16 (pre-transposed).
// Grid: (rowsA/128, cols/64) -- sized exactly, so EXEC is all-ones throughout.
template <bool RELU_BF16_OUT>
__global__ __launch_bounds__(256)
void wmma_gemm_kernel(const unsigned short* __restrict__ A,
                      const unsigned short* __restrict__ Bt,
                      const float* __restrict__ bias,
                      void* __restrict__ Out,
                      int Kdim, int cols) {
    constexpr int CTILES = 4;                    // 64 cols per block
    __shared__ __align__(16) unsigned short bSh[64 * RSTRIDE];

    const int  tid  = threadIdx.x;
    const int  lane = tid & 31;
    const int  wv   = tid >> 5;                  // wave in block, 0..7
    const int  l15  = lane & 15;
    const bool hi   = lane >= 16;

    const int tileRow = (blockIdx.x * 8 + wv) * 16;
    const int colBase = blockIdx.y * (16 * CTILES);

    // A fragment (ISA 16-bit A 16x32 layout): lanes 0-15 hold K {0..7,16..23},
    // lanes 16-31 hold K {8..15,24..31} of the same 16 rows.
    const unsigned short* aBase = A + (size_t)(tileRow + l15) * Kdim + (hi ? 8 : 0);

    v8f zero = {};
    v8f acc[CTILES];
#pragma unroll
    for (int c = 0; c < CTILES; ++c) acc[c] = zero;

    for (int k0 = 0; k0 < Kdim; k0 += KCHUNK) {
        // ---- cooperative stage of the 64-col x KCHUNK B chunk into LDS ----
        // 64 * 128 halves = 1024 x b128; 256 threads -> 4 vectors each.
#pragma unroll
        for (int i = 0; i < 4; ++i) {
            int idx = tid + i * 256;             // 0..1023
            int c   = idx >> 4;                  // column within stripe (16 vecs/row)
            int kv  = idx & 15;                  // which 8-half vector in the row
            v4u d = *(const v4u*)(Bt + (size_t)(colBase + c) * Kdim + k0 + kv * 8);
            *(v4u*)&bSh[c * RSTRIDE + kv * 8] = d;
        }
        __syncthreads();

        // ---- 4 k-steps of 32, 4 column tiles each: 16 WMMAs per chunk ----
#pragma unroll
        for (int kb = 0; kb < KCHUNK; kb += 32) {
            FragBF a;
            a.q[0] = *(const v4u*)(aBase + k0 + kb);        // K chunk 0 (8 halves)
            a.q[1] = *(const v4u*)(aBase + k0 + kb + 16);   // K chunk 1 (8 halves)
#pragma unroll
            for (int ct = 0; ct < CTILES; ++ct) {
                // B fragment: lane n = column n; lanes 0-15 K {0..7,16..23}, 16-31 K {8..15,24..31}
                const unsigned short* bp =
                    &bSh[(ct * 16 + l15) * RSTRIDE + kb + (hi ? 8 : 0)];
                FragBF b;
                b.q[0] = *(const v4u*)bp;
                b.q[1] = *(const v4u*)(bp + 16);
                acc[ct] = __builtin_amdgcn_wmma_f32_16x16x32_bf16(
                              false, a.bf, false, b.bf, (short)0, acc[ct], false, false);
            }
        }
        if (k0 + KCHUNK < Kdim)                  // global_prefetch_b8 for next A chunk
            __builtin_prefetch(aBase + k0 + KCHUNK, 0, 1);
        __syncthreads();
    }

    // ---- epilogue: bias (+relu) and store ----
#pragma unroll
    for (int ct = 0; ct < CTILES; ++ct) {
        const int   col = colBase + ct * 16 + l15;
        const float bv  = bias ? bias[col] : 0.0f;
#pragma unroll
        for (int r = 0; r < 8; ++r) {
            // C/D layout: VGPR r -> M = r (lanes 0-15) / r+8 (lanes 16-31), N = lane&15
            const int row = tileRow + r + (hi ? 8 : 0);
            float v = acc[ct][r] + bv;
            if (RELU_BF16_OUT) {
                v = v > 0.0f ? v : 0.0f;
                ((unsigned short*)Out)[(size_t)row * cols + col] = f2bf(v);
            } else {
                ((float*)Out)[(size_t)row * cols + col] = v;
            }
        }
    }
}

// --------------------------- graph structure -------------------------------

__global__ void degree_kernel(const int* __restrict__ rows, float* __restrict__ deg, long E) {
    long e = (long)blockIdx.x * blockDim.x + threadIdx.x;
    if (e < E) atomicAdd(&deg[rows[e]], 1.0f);
}

__global__ void dis_kernel(const float* __restrict__ deg, float* __restrict__ dis, int n) {
    int i = blockIdx.x * blockDim.x + threadIdx.x;
    if (i < n) dis[i] = rsqrtf(deg[i]);      // every node has a self-loop -> deg >= 1
}

// Y[r,:] += -(dis[r]*w*dis[c]) * V[c,:]   (L_tilde = -A_norm, scatter SpMM)
__global__ void spmm_scatter_kernel(const int* __restrict__ rows,
                                    const int* __restrict__ cols,
                                    const float* __restrict__ w,
                                    const float* __restrict__ dis,
                                    const float* __restrict__ V,
                                    float* __restrict__ Y, long E) {
    long t = (long)blockIdx.x * blockDim.x + threadIdx.x;
    if (t >= E * 16) return;
    long e = t >> 4;
    int  chunk = (int)(t & 15);
    int  r = rows[e], c = cols[e];
    float val = -dis[r] * w[e] * dis[c];
    float4 v = ((const float4*)(V + (size_t)c * OUT_DIM))[chunk];
    float* yp = Y + (size_t)r * OUT_DIM + chunk * 4;
    atomicAdd(yp + 0, val * v.x);
    atomicAdd(yp + 1, val * v.y);
    atomicAdd(yp + 2, val * v.z);
    atomicAdd(yp + 3, val * v.w);
}

// ---------------------- heat-kernel coefficients ---------------------------
// I_k(t) ~= sum_{m=0}^{9} (t/2)^(2m+k) / (m! (m+k)!)   via term recurrence
__global__ void coeff_kernel(const float* __restrict__ node_t,
                             float* __restrict__ coeffs, int n) {
    int i = blockIdx.x * blockDim.x + threadIdx.x;
    if (i >= n) return;
    float t  = node_t[i];
    t        = t > 0.0f ? t : 0.0f;          // relu(node_t)
    float et  = __expf(-t);
    float th  = 0.5f * t;
    float th2 = th * th;
    float pk  = 1.0f;                        // (t/2)^k / k!
    for (int k = 0; k <= KCHEB; ++k) {
        float term = pk, sum = 0.0f;
#pragma unroll
        for (int m = 0; m < 10; ++m) {
            sum  += term;
            term *= th2 / (float)((m + 1) * (m + k + 1));
        }
        float c = (k == 0) ? (et * sum)
                           : (2.0f * et * sum * ((k & 1) ? -1.0f : 1.0f));
        coeffs[(size_t)k * n + i] = c;
        pk *= th / (float)(k + 1);
    }
}

// diffusion = c0*h + c1*T1
__global__ void cheb_init_kernel(const float* __restrict__ h,
                                 const float* __restrict__ T1,
                                 const float* __restrict__ coeffs,
                                 float* __restrict__ diff, int n) {
    long i = (long)blockIdx.x * blockDim.x + threadIdx.x;
    long total = (long)n * OUT_DIM;
    if (i >= total) return;
    int node = (int)(i >> 6);
    diff[i] = coeffs[node] * h[i] + coeffs[(size_t)n + node] * T1[i];
}

// Tk = 2*Y - Tm2 (in place into Y); diffusion += ck*Tk
__global__ void cheb_update_kernel(float* __restrict__ Y,
                                   const float* __restrict__ Tm2,
                                   const float* __restrict__ ck,
                                   float* __restrict__ diff, int n) {
    long i = (long)blockIdx.x * blockDim.x + threadIdx.x;
    long total = (long)n * OUT_DIM;
    if (i >= total) return;
    int node = (int)(i >> 6);
    float tk = 2.0f * Y[i] - Tm2[i];
    Y[i] = tk;
    diff[i] += ck[node] * tk;
}

// out = log_softmax(gamma0*h + gamma1*diffusion) over 64 features
__global__ void out_kernel(const float* __restrict__ h,
                           const float* __restrict__ diff,
                           const float* __restrict__ g0,
                           const float* __restrict__ g1,
                           float* __restrict__ out, int n) {
    int i = blockIdx.x * blockDim.x + threadIdx.x;
    if (i >= n) return;
    float a = g0[i], b = g1[i];
    const float4* hp = (const float4*)(h    + (size_t)i * OUT_DIM);
    const float4* dp = (const float4*)(diff + (size_t)i * OUT_DIM);
    float v[OUT_DIM];
    float m = -3.402823466e38f;
#pragma unroll
    for (int j = 0; j < 16; ++j) {
        float4 hv = hp[j], dv = dp[j];
        v[4 * j + 0] = a * hv.x + b * dv.x;
        v[4 * j + 1] = a * hv.y + b * dv.y;
        v[4 * j + 2] = a * hv.z + b * dv.z;
        v[4 * j + 3] = a * hv.w + b * dv.w;
        m = fmaxf(m, fmaxf(fmaxf(v[4 * j], v[4 * j + 1]), fmaxf(v[4 * j + 2], v[4 * j + 3])));
    }
    float s = 0.0f;
#pragma unroll
    for (int j = 0; j < OUT_DIM; ++j) s += __expf(v[j] - m);
    float ls = __logf(s) + m;
    float4* op = (float4*)(out + (size_t)i * OUT_DIM);
#pragma unroll
    for (int j = 0; j < 16; ++j) {
        float4 o;
        o.x = v[4 * j + 0] - ls;
        o.y = v[4 * j + 1] - ls;
        o.z = v[4 * j + 2] - ls;
        o.w = v[4 * j + 3] - ls;
        op[j] = o;
    }
}

// ------------------------------- launcher ----------------------------------

extern "C" void kernel_launch(void* const* d_in, const int* in_sizes, int n_in,
                              void* d_out, int out_size, void* d_ws, size_t ws_size,
                              hipStream_t stream) {
    const float* x      = (const float*)d_in[0];
    const int*   ei     = (const int*)d_in[1];
    const float* ew     = (const float*)d_in[2];
    const float* w1     = (const float*)d_in[3];
    const float* b1     = (const float*)d_in[4];
    const float* w2     = (const float*)d_in[5];
    const float* b2     = (const float*)d_in[6];
    const float* node_t = (const float*)d_in[7];
    const float* g0     = (const float*)d_in[8];
    const float* g1     = (const float*)d_in[9];
    float*       out    = (float*)d_out;

    const int  N = in_sizes[7];              // node count (16384)
    const long E = in_sizes[2];              // edge count (edge_weight length)
    const int* erow = ei;
    const int* ecol = ei + E;

    // -------- workspace carve-out (256B aligned slices) --------
    char* ws = (char*)d_ws;
    size_t off = 0;
    auto take = [&](size_t bytes) -> char* {
        char* p = ws + off;
        off += (bytes + 255) & ~(size_t)255;
        return p;
    };
    unsigned short* xb   = (unsigned short*)take((size_t)N * IN_DIM * 2);
    unsigned short* w1t  = (unsigned short*)take((size_t)HIDDEN * IN_DIM * 2);
    unsigned short* w2t  = (unsigned short*)take((size_t)OUT_DIM * HIDDEN * 2);
    unsigned short* h1b  = (unsigned short*)take((size_t)N * HIDDEN * 2);
    float* hf   = (float*)take((size_t)N * OUT_DIM * 4);
    float* bufA = (float*)take((size_t)N * OUT_DIM * 4);
    float* bufB = (float*)take((size_t)N * OUT_DIM * 4);
    float* bufC = (float*)take((size_t)N * OUT_DIM * 4);
    float* diff = (float*)take((size_t)N * OUT_DIM * 4);
    float* deg  = (float*)take((size_t)N * 4);
    float* dis  = (float*)take((size_t)N * 4);
    float* cofs = (float*)take((size_t)(KCHEB + 1) * N * 4);

    const int T = 256;
    auto blocks = [](long n, int t) { return (unsigned)((n + t - 1) / t); };

    // -------- precision conversion / weight transpose --------
    cvt_bf16_kernel<<<blocks((long)N * IN_DIM, T), T, 0, stream>>>(x, xb, (long)N * IN_DIM);
    transpose_cvt_kernel<<<blocks((long)IN_DIM * HIDDEN, T), T, 0, stream>>>(w1, w1t, IN_DIM, HIDDEN);
    transpose_cvt_kernel<<<blocks((long)HIDDEN * OUT_DIM, T), T, 0, stream>>>(w2, w2t, HIDDEN, OUT_DIM);

    // -------- MLP via WMMA (LDS-staged weight stripes) --------
    // GEMM1: [N x 512] @ [512 x 256] -> relu -> bf16
    wmma_gemm_kernel<true><<<dim3(N / 128, HIDDEN / 64), T, 0, stream>>>(
        xb, w1t, b1, h1b, IN_DIM, HIDDEN);
    // GEMM2: [N x 256] @ [256 x 64] -> f32
    wmma_gemm_kernel<false><<<dim3(N / 128, OUT_DIM / 64), T, 0, stream>>>(
        h1b, w2t, b2, hf, HIDDEN, OUT_DIM);

    // -------- graph normalization --------
    zero_f32_kernel<<<blocks(N, T), T, 0, stream>>>(deg, N);
    degree_kernel<<<blocks(E, T), T, 0, stream>>>(erow, deg, E);
    dis_kernel<<<blocks(N, T), T, 0, stream>>>(deg, dis, N);

    // -------- heat-kernel coefficients --------
    coeff_kernel<<<blocks(N, T), T, 0, stream>>>(node_t, cofs, N);

    // -------- Chebyshev recursion (sparse L_tilde) --------
    const long NF = (long)N * OUT_DIM;
    // T1 = L_tilde @ h
    zero_f32_kernel<<<blocks(NF, T), T, 0, stream>>>(bufA, NF);
    spmm_scatter_kernel<<<blocks(E * 16, T), T, 0, stream>>>(erow, ecol, ew, dis, hf, bufA, E);
    // diffusion = c0*h + c1*T1
    cheb_init_kernel<<<blocks(NF, T), T, 0, stream>>>(hf, bufA, cofs, diff, N);

    float* bufs[3] = {bufA, bufB, bufC};
    const float* prev2 = hf;                 // T0
    float*       prev1 = bufA;               // T1
    for (int k = 2; k <= KCHEB; ++k) {
        float* cur = bufs[(k - 1) % 3];
        zero_f32_kernel<<<blocks(NF, T), T, 0, stream>>>(cur, NF);
        spmm_scatter_kernel<<<blocks(E * 16, T), T, 0, stream>>>(erow, ecol, ew, dis, prev1, cur, E);
        cheb_update_kernel<<<blocks(NF, T), T, 0, stream>>>(cur, prev2, cofs + (size_t)k * N, diff, N);
        prev2 = prev1;
        prev1 = cur;
    }

    // -------- combine + log_softmax --------
    out_kernel<<<blocks(N, T), T, 0, stream>>>(hf, diff, g0, g1, out, N);
}
